// TransformerBlock_13211319403308
// MI455X (gfx1250) — compile-verified, matrix operand order checked by reading, not run on hardware
//
#include <hip/hip_runtime.h>

// ---------------- problem constants ----------------
#define kB   16
#define kS   1024
#define kD   512
#define kH   8
#define kE   8
#define kF   2048
#define kDK  64
#define kN   (kB * kS)          // 16384
#define kCAP 2560               // int(1.25*N/E)
#define kNB  (kN / 256)         // 64 router blocks

// output layout (floats): out[N*D], counts[E], route_prob[E], n_dropped[1], rpm[N]
#define OFF_COUNTS ((size_t)kN * kD)
#define OFF_RPROB  (OFF_COUNTS + kE)
#define OFF_NDROP  (OFF_COUNTS + 2 * kE)
#define OFF_RPM    (OFF_COUNTS + 2 * kE + 1)

// ---------------- types ----------------
typedef __attribute__((ext_vector_type(16))) __bf16 v16bf;
typedef __attribute__((ext_vector_type(8)))  float  v8f;
typedef __attribute__((ext_vector_type(4)))  unsigned int v4u;
typedef __attribute__((ext_vector_type(8)))  int  v8i;
typedef __attribute__((ext_vector_type(4)))  int  v4i;

union BFrag {
  v16bf v;
  unsigned short u[16];
  uint4 q[2];
};

__device__ inline unsigned short f2bf(float f) {
  unsigned int u = __float_as_uint(f);
  u += 0x7FFFu + ((u >> 16) & 1u);   // round-to-nearest-even
  return (unsigned short)(u >> 16);
}

// K index inside a 16-bit A/B fragment (ISA 16-bit A-matrix 16x32 layout)
__device__ inline int kidx(int i, int hf) {
  return (i < 8) ? (hf * 8 + i) : (16 + hf * 8 + (i - 8));
}

#if defined(__has_builtin)
#if __has_builtin(__builtin_amdgcn_tensor_load_to_lds)
#define HAVE_TDM 1
#endif
#endif

#ifdef HAVE_TDM
// Issue one TDM 2D tile load: tile_k x tile_m bf16 elements, row stride
// ldK elements, global->LDS (row-major, rows packed consecutively).
// Must be executed by exactly one wave; tracked with TENSORcnt.
__device__ inline void tdm_load_a(const unsigned short* gtile, unsigned lds_off,
                                  int tile_k, int tile_m, int ldK) {
  const unsigned long long ga = (unsigned long long)gtile;
  v4u g0;
  g0.x = 1u;                                         // count=1, user mode
  g0.y = lds_off;                                    // lds_addr (bytes)
  g0.z = (unsigned)(ga & 0xFFFFFFFFu);               // global_addr[31:0]
  g0.w = (unsigned)((ga >> 32) & 0x1FFFFFFu) | (2u << 30);  // addr[56:32] | type=2
  v8i g1;
  g1[0] = (1 << 16);                                 // data_size=1 (2 bytes)
  g1[1] = (tile_k & 0xFFFF) << 16;                   // tensor_dim0[15:0] (=tile_k)
  g1[2] = ((unsigned)tile_k >> 16) | ((tile_m & 0xFFFF) << 16);  // dim0 hi | dim1 lo
  g1[3] = ((unsigned)tile_m >> 16) | (tile_k << 16); // dim1 hi | tile_dim0
  g1[4] = tile_m;                                    // tile_dim1 | tile_dim2=0
  g1[5] = ldK;                                       // tensor_dim0_stride[31:0]
  g1[6] = 0;                                         // stride0 hi | stride1 lo
  g1[7] = 0;
  v4i z4; z4[0] = 0; z4[1] = 0; z4[2] = 0; z4[3] = 0;
#if __clang_major__ >= 23
  v8i z8; z8[0]=0; z8[1]=0; z8[2]=0; z8[3]=0; z8[4]=0; z8[5]=0; z8[6]=0; z8[7]=0;
  __builtin_amdgcn_tensor_load_to_lds(g0, g1, z4, z4, z8, 0);
#else
  __builtin_amdgcn_tensor_load_to_lds(g0, g1, z4, z4, 0);
#endif
}
__device__ inline unsigned lds_off_of(const void* p) {
  // LDS flat aperture keeps the byte offset in the low 32 bits
  return (unsigned)(unsigned long long)p;
}
#endif

// ---------------- fp32 -> bf16 convert ----------------
__global__ void cvt_kernel(const float* __restrict__ in,
                           unsigned short* __restrict__ out, long n) {
  long i = (long)blockIdx.x * blockDim.x + threadIdx.x;
  if (i < n) out[i] = f2bf(in[i]);
}

// stage 32x128 B tile (row-major global) transposed into LDS [n][k]
__device__ inline void stage_B_tile(const unsigned short* __restrict__ Bm,
                                    unsigned short* __restrict__ lB,
                                    int k0, int Nc, int n0, int br, int bc) {
  const uint4 b0 = *(const uint4*)(Bm + (long)(k0 + br) * Nc + n0 + bc);
  const uint4 b1 = *(const uint4*)(Bm + (long)(k0 + br) * Nc + n0 + bc + 8);
  const unsigned short* p0 = (const unsigned short*)&b0;
  const unsigned short* p1 = (const unsigned short*)&b1;
#pragma unroll
  for (int i = 0; i < 8; ++i) lB[(bc + i) * 32 + br] = p0[i];
#pragma unroll
  for (int i = 0; i < 8; ++i) lB[(bc + 8 + i) * 32 + br] = p1[i];
}

// ---------------- tiled WMMA GEMM  C[M,Nc] = A[M,K]*B[K,Nc] + bias ----------
// OUTMODE: 0 = fp32 out, 1 = bf16 out, 2 = bf16 out + ReLU
// Requires M%64==0, Nc%128==0, K%32==0.
// Double-buffered: A tile via Tensor Data Mover (TENSORcnt), B tile via
// VGPR-transpose staging; one barrier per K-step.
template <int OUTMODE>
__global__ __launch_bounds__(256) void gemm_bf16(
    const unsigned short* __restrict__ A, long long sA,
    const unsigned short* __restrict__ Bm, long long sB,
    const float* __restrict__ bias, long long sBias,
    void* __restrict__ Cout, long long sC,
    int M, int Nc, int K) {
  __shared__ unsigned short lA[2][64 * 32];    // [m][k]
  __shared__ unsigned short lB[2][128 * 32];   // transposed: [n][k]
  const int z = blockIdx.z;
  A += z * sA; Bm += z * sB; bias += z * sBias;
  const int m0 = blockIdx.y * 64;
  const int n0 = blockIdx.x * 128;
  const int tid = threadIdx.x;
  const int lane = tid & 31, wave = tid >> 5;
  const int wm = (wave >> 2) * 32, wn = (wave & 3) * 32;
  const int hf = lane >> 4, mr = lane & 15;
  const int ar = tid >> 2, ac = (tid & 3) * 8;     // A: 64 rows x 32 cols
  const int br = tid >> 3, bc = (tid & 7) * 16;    // B: 32 rows x 128 cols
  v8f acc[2][2] = {};
  const int nK = K >> 5;
  // ---- prologue: fill buffer 0 ----
  stage_B_tile(Bm, lB[0], 0, Nc, n0, br, bc);
#ifdef HAVE_TDM
  if (wave == 0) {
    tdm_load_a(A + (long)m0 * K, lds_off_of(&lA[0][0]), 32, 64, K);
    __builtin_amdgcn_s_wait_tensorcnt(0);
  }
#else
  *(uint4*)(&lA[0][ar * 32 + ac]) = *(const uint4*)(A + (long)(m0 + ar) * K + ac);
#endif
  __syncthreads();
  for (int kt = 0; kt < nK; ++kt) {
    const int buf = kt & 1;
    // fragments from current buffer
    BFrag a[2], b[2];
#pragma unroll
    for (int t = 0; t < 2; ++t) {
      const int rowa = wm + t * 16 + mr;
      a[t].q[0] = *(const uint4*)(&lA[buf][rowa * 32 + hf * 8]);
      a[t].q[1] = *(const uint4*)(&lA[buf][rowa * 32 + 16 + hf * 8]);
      const int coln = wn + t * 16 + mr;
      b[t].q[0] = *(const uint4*)(&lB[buf][coln * 32 + hf * 8]);
      b[t].q[1] = *(const uint4*)(&lB[buf][coln * 32 + 16 + hf * 8]);
    }
    // stage next buffer while computing
    const bool more = (kt + 1) < nK;
    if (more) {
      const int k1 = (kt + 1) * 32;
      stage_B_tile(Bm, lB[buf ^ 1], k1, Nc, n0, br, bc);
#ifdef HAVE_TDM
      if (wave == 0)
        tdm_load_a(A + (long)m0 * K + k1, lds_off_of(&lA[buf ^ 1][0]), 32, 64, K);
#else
      *(uint4*)(&lA[buf ^ 1][ar * 32 + ac]) =
          *(const uint4*)(A + (long)(m0 + ar) * K + k1 + ac);
#endif
    }
#pragma unroll
    for (int i = 0; i < 2; ++i)
#pragma unroll
      for (int j = 0; j < 2; ++j)
        acc[i][j] = __builtin_amdgcn_wmma_f32_16x16x32_bf16(
            false, a[i].v, false, b[j].v, (short)0, acc[i][j], false, false);
#ifdef HAVE_TDM
    if (more && wave == 0) __builtin_amdgcn_s_wait_tensorcnt(0);
#endif
    __syncthreads();
  }
  // store C tile (+bias, opt relu/bf16)
#pragma unroll
  for (int i = 0; i < 2; ++i) {
#pragma unroll
    for (int j = 0; j < 2; ++j) {
      const int col = n0 + wn + j * 16 + mr;
      const float bv = bias[col];
#pragma unroll
      for (int v = 0; v < 8; ++v) {
        const int row = m0 + wm + i * 16 + hf * 8 + v;
        float r = acc[i][j][v] + bv;
        if (OUTMODE == 2) r = r > 0.f ? r : 0.f;
        if (OUTMODE == 0)
          ((float*)Cout)[z * sC + (long)row * Nc + col] = r;
        else
          ((unsigned short*)Cout)[z * sC + (long)row * Nc + col] = f2bf(r);
      }
    }
  }
}

// ---------------- attention: one wave handles 16 query rows ----------------
// q,k,v bf16 laid out [B,S,H,DK] (== [N,D] row-major); ctx bf16 same layout.
__global__ __launch_bounds__(32) void attn_kernel(
    const unsigned short* __restrict__ q,
    const unsigned short* __restrict__ k,
    const unsigned short* __restrict__ v,
    unsigned short* __restrict__ ctx) {
  __shared__ float sc[16 * kS];   // 64 KB score buffer
  const int qt = blockIdx.x, h = blockIdx.y, b = blockIdx.z;
  const int lane = threadIdx.x & 31;
  const int hf = lane >> 4, mr = lane & 15, nr = lane & 15;
  const long base = ((long)b * kS) * kD + h * kDK;
  const int q0 = qt * 16;
  // hoist Q fragments (DK=64 -> two K=32 fragments)
  BFrag qa[2];
#pragma unroll
  for (int kk = 0; kk < 2; ++kk) {
    const unsigned short* qp = q + base + (long)(q0 + mr) * kD + kk * 32;
    qa[kk].q[0] = *(const uint4*)(qp + hf * 8);
    qa[kk].q[1] = *(const uint4*)(qp + 16 + hf * 8);
  }
  const float scale = 0.125f;   // 1/sqrt(64)
  for (int kt = 0; kt < kS / 16; ++kt) {
    v8f acc = {};
#pragma unroll
    for (int kk = 0; kk < 2; ++kk) {
      BFrag kb;
      const unsigned short* kp = k + base + (long)(kt * 16 + nr) * kD + kk * 32;
      kb.q[0] = *(const uint4*)(kp + hf * 8);
      kb.q[1] = *(const uint4*)(kp + 16 + hf * 8);
      acc = __builtin_amdgcn_wmma_f32_16x16x32_bf16(
          false, qa[kk].v, false, kb.v, (short)0, acc, false, false);
    }
#pragma unroll
    for (int vv = 0; vv < 8; ++vv)
      sc[(vv + hf * 8) * kS + kt * 16 + nr] = acc[vv] * scale;
  }
  __syncthreads();
  // softmax: 2 lanes per row, half-wave reduction
  {
    float* row = sc + mr * kS + hf * (kS / 2);
    float mx = -3.4e38f;
    for (int j = 0; j < kS / 2; ++j) mx = fmaxf(mx, row[j]);
    mx = fmaxf(mx, __shfl_xor(mx, 16, 32));
    float sum = 0.f;
    for (int j = 0; j < kS / 2; ++j) { float e = __expf(row[j] - mx); row[j] = e; sum += e; }
    sum += __shfl_xor(sum, 16, 32);
    const float inv = 1.f / sum;
    for (int j = 0; j < kS / 2; ++j) row[j] *= inv;
  }
  __syncthreads();
  // ctx = P (16 x S) x V (S x 64)
  v8f co[4] = {};
  for (int kc = 0; kc < kS / 32; ++kc) {
    BFrag pa;
#pragma unroll
    for (int i = 0; i < 16; ++i)
      pa.u[i] = f2bf(sc[mr * kS + kc * 32 + kidx(i, hf)]);
#pragma unroll
    for (int ct = 0; ct < 4; ++ct) {
      BFrag vb;
#pragma unroll
      for (int i = 0; i < 16; ++i)
        vb.u[i] = v[base + (long)(kc * 32 + kidx(i, hf)) * kD + ct * 16 + nr];
      co[ct] = __builtin_amdgcn_wmma_f32_16x16x32_bf16(
          false, pa.v, false, vb.v, (short)0, co[ct], false, false);
    }
  }
#pragma unroll
  for (int ct = 0; ct < 4; ++ct)
#pragma unroll
    for (int vv = 0; vv < 8; ++vv)
      ctx[base + (long)(q0 + vv + hf * 8) * kD + ct * 16 + nr] = f2bf(co[ct][vv]);
}

// ---------------- residual + LayerNorm (x + r), fp32 + bf16 outputs --------
__global__ __launch_bounds__(256) void resid_ln_kernel(
    const float* __restrict__ x, const float* __restrict__ r,
    const float* __restrict__ g, const float* __restrict__ bta,
    float* __restrict__ outf, unsigned short* __restrict__ outb) {
  __shared__ float red[256];
  const int n = blockIdx.x, t = threadIdx.x;
  const long xb = (long)n * kD;
  float v0 = x[xb + t] + r[xb + t];
  float v1 = x[xb + 256 + t] + r[xb + 256 + t];
  red[t] = v0 + v1; __syncthreads();
  for (int o = 128; o > 0; o >>= 1) { if (t < o) red[t] += red[t + o]; __syncthreads(); }
  const float mean = red[0] / (float)kD; __syncthreads();
  const float d0 = v0 - mean, d1 = v1 - mean;
  red[t] = d0 * d0 + d1 * d1; __syncthreads();
  for (int o = 128; o > 0; o >>= 1) { if (t < o) red[t] += red[t + o]; __syncthreads(); }
  const float inv = rsqrtf(red[0] / (float)kD + 1e-5f);
  const float o0 = d0 * inv * g[t] + bta[t];
  const float o1 = d1 * inv * g[256 + t] + bta[256 + t];
  outf[xb + t] = o0; outf[xb + 256 + t] = o1;
  outb[xb + t] = f2bf(o0); outb[xb + 256 + t] = f2bf(o1);
}

// ---------------- router: logits, softmax, top-1, block counts -------------
__global__ __launch_bounds__(256) void router_kernel(
    const float* __restrict__ x1, const float* __restrict__ Wr,
    const float* __restrict__ br,
    int* __restrict__ routes, int* __restrict__ rank, float* __restrict__ rpm,
    int* __restrict__ blockcnt, float* __restrict__ countsF,
    float* __restrict__ rprob) {
  __shared__ int sroute[256];
  const int n = blockIdx.x * 256 + threadIdx.x;
  float logit[kE];
#pragma unroll
  for (int e = 0; e < kE; ++e) logit[e] = br[e];
  const float* xr = x1 + (long)n * kD;
  for (int d = 0; d < kD; ++d) {
    const float xv = xr[d];
#pragma unroll
    for (int e = 0; e < kE; ++e) logit[e] += xv * Wr[d * kE + e];
  }
  float mx = logit[0];
#pragma unroll
  for (int e = 1; e < kE; ++e) mx = fmaxf(mx, logit[e]);
  float sum = 0.f;
#pragma unroll
  for (int e = 0; e < kE; ++e) { logit[e] = __expf(logit[e] - mx); sum += logit[e]; }
  const float inv = 1.f / sum;
  int best = 0; float bp = logit[0] * inv;
#pragma unroll
  for (int e = 1; e < kE; ++e) {
    const float p = logit[e] * inv;
    if (p > bp) { bp = p; best = e; }
  }
#pragma unroll
  for (int e = 0; e < kE; ++e) atomicAdd(&rprob[e], logit[e] * inv);
  atomicAdd(&countsF[best], 1.0f);
  routes[n] = best; rpm[n] = bp;
  sroute[threadIdx.x] = best; __syncthreads();
  int rk = 0;
  for (int j = 0; j < threadIdx.x; ++j) rk += (sroute[j] == best) ? 1 : 0;
  rank[n] = rk;
  if (threadIdx.x < kE) {
    int c = 0;
    for (int j = 0; j < 256; ++j) c += (sroute[j] == threadIdx.x) ? 1 : 0;
    blockcnt[blockIdx.x * kE + threadIdx.x] = c;
  }
}

// exclusive prefix scan of per-block expert counts (tiny)
__global__ void scan_kernel(const int* __restrict__ blockcnt,
                            int* __restrict__ blockbase) {
  const int e = threadIdx.x;
  if (e < kE) {
    int run = 0;
    for (int b = 0; b < kNB; ++b) { blockbase[b * kE + e] = run; run += blockcnt[b * kE + e]; }
  }
}

// ---------------- dispatch: scatter kept tokens into [E,CAP,D] -------------
__global__ __launch_bounds__(64) void dispatch_kernel(
    const int* __restrict__ routes, const int* __restrict__ rank,
    const int* __restrict__ blockbase, int* __restrict__ pos,
    const unsigned short* __restrict__ x1b,
    unsigned short* __restrict__ disp, int* __restrict__ ndrop) {
  __shared__ int sh[3];
  const int n = blockIdx.x;
  if (threadIdx.x == 0) {
    const int e = routes[n];
    const int p = blockbase[(n >> 8) * kE + e] + rank[n];
    pos[n] = p;
    const int keep = p < kCAP;
    if (!keep) atomicAdd(ndrop, 1);
    sh[0] = e; sh[1] = keep ? p : (kCAP - 1); sh[2] = keep;
  }
  __syncthreads();
  if (sh[2]) {
    const unsigned short* src = x1b + (long)n * kD;
    unsigned short* dst = disp + ((long)sh[0] * kCAP + sh[1]) * kD;
    const int d = threadIdx.x * 8;
    *(uint4*)(dst + d) = *(const uint4*)(src + d);
  }
}

// ---------------- combine + residual + LayerNorm2 --------------------------
__global__ __launch_bounds__(256) void combine_ln_kernel(
    const float* __restrict__ x1, const float* __restrict__ eo,
    const int* __restrict__ routes, const int* __restrict__ pos,
    const float* __restrict__ rpm,
    const float* __restrict__ g, const float* __restrict__ bta,
    float* __restrict__ out, float* __restrict__ rpm_out) {
  __shared__ float red[256];
  const int n = blockIdx.x, t = threadIdx.x;
  const int e = routes[n], p = pos[n];
  const bool keep = p < kCAP;
  const int pc = keep ? p : (kCAP - 1);
  const float w = rpm[n];
  const long xb = (long)n * kD;
  const long eb = ((long)e * kCAP + pc) * kD;
  const float y0 = (keep ? eo[eb + t]       : x1[xb + t])       * w;
  const float y1 = (keep ? eo[eb + 256 + t] : x1[xb + 256 + t]) * w;
  float v0 = x1[xb + t] + y0;
  float v1 = x1[xb + 256 + t] + y1;
  red[t] = v0 + v1; __syncthreads();
  for (int o = 128; o > 0; o >>= 1) { if (t < o) red[t] += red[t + o]; __syncthreads(); }
  const float mean = red[0] / (float)kD; __syncthreads();
  const float d0 = v0 - mean, d1 = v1 - mean;
  red[t] = d0 * d0 + d1 * d1; __syncthreads();
  for (int o = 128; o > 0; o >>= 1) { if (t < o) red[t] += red[t + o]; __syncthreads(); }
  const float inv = rsqrtf(red[0] / (float)kD + 1e-5f);
  out[xb + t]       = d0 * inv * g[t] + bta[t];
  out[xb + 256 + t] = d1 * inv * g[256 + t] + bta[256 + t];
  if (t == 0) rpm_out[n] = w;
}

// ---------------- tiny finalize -------------------------------------------
__global__ void finalize_kernel(const float* __restrict__ countsF,
                                const float* __restrict__ rprob,
                                const int* __restrict__ ndrop,
                                float* __restrict__ out) {
  const int t = threadIdx.x;
  if (t < kE) out[OFF_COUNTS + t] = countsF[t];
  else if (t < 2 * kE) out[OFF_RPROB + (t - kE)] = rprob[t - kE];
  else if (t == 2 * kE) out[OFF_NDROP] = (float)(*ndrop);
}

// ---------------- host launcher -------------------------------------------
extern "C" void kernel_launch(void* const* d_in, const int* in_sizes, int n_in,
                              void* d_out, int out_size, void* d_ws, size_t ws_size,
                              hipStream_t stream) {
  const float* x    = (const float*)d_in[0];
  const float* Wq   = (const float*)d_in[1];
  const float* bq   = (const float*)d_in[2];
  const float* Wk   = (const float*)d_in[3];
  const float* bk   = (const float*)d_in[4];
  const float* Wv   = (const float*)d_in[5];
  const float* bv   = (const float*)d_in[6];
  const float* Wo   = (const float*)d_in[7];
  const float* bo   = (const float*)d_in[8];
  const float* ln1g = (const float*)d_in[9];
  const float* ln1b = (const float*)d_in[10];
  const float* Wr   = (const float*)d_in[11];
  const float* br   = (const float*)d_in[12];
  const float* W1   = (const float*)d_in[13];
  const float* b1   = (const float*)d_in[14];
  const float* W2   = (const float*)d_in[15];
  const float* b2   = (const float*)d_in[16];
  const float* ln2g = (const float*)d_in[17];
  const float* ln2b = (const float*)d_in[18];
  float* outF = (float*)d_out;

  char* ws = (char*)d_ws;
  size_t off = 0;
  auto take = [&](size_t bytes) -> char* {
    char* p = ws + off;
    off += (bytes + 255) & ~(size_t)255;
    return p;
  };
  unsigned short* xbf   = (unsigned short*)take((size_t)kN * kD * 2);
  unsigned short* wqbf  = (unsigned short*)take((size_t)kD * kD * 2);
  unsigned short* wkbf  = (unsigned short*)take((size_t)kD * kD * 2);
  unsigned short* wvbf  = (unsigned short*)take((size_t)kD * kD * 2);
  unsigned short* wobf  = (unsigned short*)take((size_t)kD * kD * 2);
  unsigned short* w1bf  = (unsigned short*)take((size_t)kE * kD * kF * 2);
  unsigned short* w2bf  = (unsigned short*)take((size_t)kE * kF * kD * 2);
  unsigned short* qbf   = (unsigned short*)take((size_t)kN * kD * 2);
  unsigned short* kbf   = (unsigned short*)take((size_t)kN * kD * 2);
  unsigned short* vbf   = (unsigned short*)take((size_t)kN * kD * 2);
  unsigned short* ctxbf = (unsigned short*)take((size_t)kN * kD * 2);
  float*          attnf = (float*)take((size_t)kN * kD * 4);
  float*          x1f   = (float*)take((size_t)kN * kD * 4);
  unsigned short* x1bf  = (unsigned short*)take((size_t)kN * kD * 2);
  unsigned short* dispb = (unsigned short*)take((size_t)kE * kCAP * kD * 2);
  unsigned short* hbf   = (unsigned short*)take((size_t)kE * kCAP * kF * 2);
  float*          eof   = (float*)take((size_t)kE * kCAP * kD * 4);
  int*   routes   = (int*)take((size_t)kN * 4);
  int*   rankArr  = (int*)take((size_t)kN * 4);
  int*   posArr   = (int*)take((size_t)kN * 4);
  float* rpmW     = (float*)take((size_t)kN * 4);
  int*   blockcnt = (int*)take((size_t)kNB * kE * 4);
  int*   blockbase= (int*)take((size_t)kNB * kE * 4);
  float* countsF  = (float*)take(256);       // + rprob + ndrop in same zone
  float* rprob    = countsF + kE;
  int*   ndrop    = (int*)(countsF + 2 * kE);

  const dim3 blk256(256);
  // --- bf16 conversions ---
  cvt_kernel<<<((long)kN * kD + 255) / 256, blk256, 0, stream>>>(x, xbf, (long)kN * kD);
  cvt_kernel<<<((long)kD * kD + 255) / 256, blk256, 0, stream>>>(Wq, wqbf, (long)kD * kD);
  cvt_kernel<<<((long)kD * kD + 255) / 256, blk256, 0, stream>>>(Wk, wkbf, (long)kD * kD);
  cvt_kernel<<<((long)kD * kD + 255) / 256, blk256, 0, stream>>>(Wv, wvbf, (long)kD * kD);
  cvt_kernel<<<((long)kD * kD + 255) / 256, blk256, 0, stream>>>(Wo, wobf, (long)kD * kD);
  cvt_kernel<<<((long)kE * kD * kF + 255) / 256, blk256, 0, stream>>>(W1, w1bf, (long)kE * kD * kF);
  cvt_kernel<<<((long)kE * kF * kD + 255) / 256, blk256, 0, stream>>>(W2, w2bf, (long)kE * kF * kD);
  // --- QKV projections (bf16 out) ---
  gemm_bf16<1><<<dim3(kD / 128, kN / 64, 1), blk256, 0, stream>>>(
      xbf, 0, wqbf, 0, bq, 0, qbf, 0, kN, kD, kD);
  gemm_bf16<1><<<dim3(kD / 128, kN / 64, 1), blk256, 0, stream>>>(
      xbf, 0, wkbf, 0, bk, 0, kbf, 0, kN, kD, kD);
  gemm_bf16<1><<<dim3(kD / 128, kN / 64, 1), blk256, 0, stream>>>(
      xbf, 0, wvbf, 0, bv, 0, vbf, 0, kN, kD, kD);
  // --- attention ---
  attn_kernel<<<dim3(kS / 16, kH, kB), dim3(32), 0, stream>>>(qbf, kbf, vbf, ctxbf);
  // --- output projection (fp32 out) ---
  gemm_bf16<0><<<dim3(kD / 128, kN / 64, 1), blk256, 0, stream>>>(
      ctxbf, 0, wobf, 0, bo, 0, attnf, 0, kN, kD, kD);
  // --- residual + LN1 ---
  resid_ln_kernel<<<kN, blk256, 0, stream>>>(x, attnf, ln1g, ln1b, x1f, x1bf);
  // --- routing ---
  hipMemsetAsync(countsF, 0, 256, stream);
  router_kernel<<<kNB, blk256, 0, stream>>>(x1f, Wr, br, routes, rankArr, rpmW,
                                            blockcnt, countsF, rprob);
  scan_kernel<<<1, 32, 0, stream>>>(blockcnt, blockbase);
  dispatch_kernel<<<kN, 64, 0, stream>>>(routes, rankArr, blockbase, posArr,
                                         x1bf, dispb, ndrop);
  // --- expert FFN (batched over experts via grid.z) ---
  gemm_bf16<2><<<dim3(kF / 128, kCAP / 64, kE), blk256, 0, stream>>>(
      dispb, (long long)kCAP * kD, w1bf, (long long)kD * kF, b1, kF,
      hbf, (long long)kCAP * kF, kCAP, kF, kD);
  gemm_bf16<0><<<dim3(kD / 128, kCAP / 64, kE), blk256, 0, stream>>>(
      hbf, (long long)kCAP * kF, w2bf, (long long)kF * kD, b2, kD,
      eof, (long long)kCAP * kD, kCAP, kD, kF);
  // --- combine + residual + LN2 + rpm output ---
  combine_ln_kernel<<<kN, blk256, 0, stream>>>(x1f, eof, routes, posArr, rpmW,
                                               ln2g, ln2b, outF, outF + OFF_RPM);
  finalize_kernel<<<1, 32, 0, stream>>>(countsF, rprob, ndrop, outF);
}
